// EGHN_38448547234244
// MI455X (gfx1250) — matrix operand order repaired and checked by reading, not compiled, not run on hardware
//
#include <hip/hip_runtime.h>
#include <hip/hip_bf16.h>

// ---------------------------------------------------------------------------
// EGHN forward on gfx1250: edge/node MLPs via v_wmma_f32_16x16x32_f16,
// f32 accumulation, f16 staged operands, pre-swizzled weight fragments.
// ---------------------------------------------------------------------------

typedef __attribute__((ext_vector_type(16))) _Float16 v16h;
typedef __attribute__((ext_vector_type(8)))  float    v8f;

#define NN   20000
#define EE   640000
#define HDIM 128
#define KPAD_MSG 288   // 267 padded to 9 * 32
#define TILE 16

// ---------------- small helpers ----------------

__device__ __forceinline__ unsigned short f2h(float x) {
    _Float16 h = (_Float16)x;
    unsigned short u;
    __builtin_memcpy(&u, &h, 2);
    return u;
}

__device__ __forceinline__ float h2f(unsigned short u) {
    _Float16 h;
    __builtin_memcpy(&h, &u, 2);
    return (float)h;
}

__device__ __forceinline__ unsigned pack2h(float a, float b) {
    return (unsigned)f2h(a) | ((unsigned)f2h(b) << 16);
}

__device__ __forceinline__ float silu_f(float x) {
    return x / (1.0f + __expf(-x));
}

// A fragment (16x32 f16) from row-major f16 tile in LDS.
// ISA layout: lane = M row (lane&15); lanes<16 hold K {0..7,16..23} of the
// 32-chunk, lanes>=16 hold K {8..15,24..31}; VGPR j packs 2 consecutive K.
__device__ __forceinline__ v16h loadAfrag(const unsigned short* a, int pitch,
                                          int kc, int lane) {
    union { unsigned u[8]; v16h h; } r;
    const int m    = lane & 15;
    const int koff = (lane < 16) ? 0 : 8;
    const unsigned short* rowp = a + m * pitch + kc * 32 + koff;
#pragma unroll
    for (int j = 0; j < 8; ++j) {
        const int k = ((j >= 4) ? 16 : 0) + 2 * (j & 3);
        r.u[j] = *(const unsigned*)(rowp + k);   // 4B-aligned (pitch, k even)
    }
    return r.h;
}

// Same A fragment but sourced from an f32 tile (convert on the fly).
__device__ __forceinline__ v16h loadAfrag_f32(const float* a, int pitch,
                                              int kc, int lane) {
    union { unsigned u[8]; v16h h; } r;
    const int m    = lane & 15;
    const int koff = (lane < 16) ? 0 : 8;
    const float* rowp = a + m * pitch + kc * 32 + koff;
#pragma unroll
    for (int j = 0; j < 8; ++j) {
        const int k = ((j >= 4) ? 16 : 0) + 2 * (j & 3);
        r.u[j] = pack2h(rowp[k], rowp[k + 1]);
    }
    return r.h;
}

// B fragment from pre-swizzled global weights: 8 contiguous dwords per lane.
__device__ __forceinline__ v16h loadBfrag(const unsigned* __restrict__ w,
                                          int ntiles, int kc, int nt, int lane) {
    union { unsigned u[8]; v16h h; } r;
    const unsigned* p = w + (((size_t)(kc * ntiles + nt) * 32 + lane) * 8);
#pragma unroll
    for (int j = 0; j < 8; ++j) r.u[j] = p[j];
    return r.h;
}

// C/D layout: lane = N col (lane&15); VGPR j holds M = j + (lane<16 ? 0 : 8).
__device__ __forceinline__ void storeC_h16(unsigned short* dst, int pitch,
                                           int ncol, int lane, v8f c,
                                           const float* __restrict__ bias,
                                           bool act) {
    const int n     = ncol + (lane & 15);
    const float bv  = bias[n];
    const int mbase = (lane < 16) ? 0 : 8;
#pragma unroll
    for (int j = 0; j < 8; ++j) {
        float v = c[j] + bv;
        if (act) v = silu_f(v);
        dst[(mbase + j) * pitch + n] = f2h(v);
    }
}

__device__ __forceinline__ void storeC_f32(float* dst, int pitch, int ncol,
                                           int lane, v8f c,
                                           const float* __restrict__ bias,
                                           bool act) {
    const int n     = ncol + (lane & 15);
    const float bv  = bias[n];
    const int mbase = (lane < 16) ? 0 : 8;
#pragma unroll
    for (int j = 0; j < 8; ++j) {
        float v = c[j] + bv;
        if (act) v = silu_f(v);
        dst[(mbase + j) * pitch + n] = v;
    }
}

// ---------------- utility kernels ----------------

__global__ void fill_f32(float* p, int n) {
    int i = blockIdx.x * blockDim.x + threadIdx.x;
    if (i < n) p[i] = 0.0f;
}

__global__ void copy_f32(const float* __restrict__ s, float* __restrict__ d, int n) {
    int i = blockIdx.x * blockDim.x + threadIdx.x;
    if (i < n) d[i] = s[i];
}

// Convert f32 row-major [Ksrc x 128] weights to f16 in the exact B-fragment
// layout: dword index = ((kt*8 + nt)*32 + lane)*8 + j,
//   n = nt*16 + (lane&15), k = kt*32 + (lane<16?0:8) + (j>=4?16:0) + 2*(j&3) (+1 in hi half).
__global__ void swizzle_w_kernel(const float* __restrict__ src, int Ksrc,
                                 int Kpad, unsigned* __restrict__ dst) {
    int id = blockIdx.x * blockDim.x + threadIdx.x;
    if (id >= Kpad * 64) return;            // Kpad/32 * 8 * 32 * 8 dwords
    const int j    = id & 7;
    const int lane = (id >> 3) & 31;
    const int nt   = (id >> 8) & 7;
    const int kt   = id >> 11;
    const int n    = nt * 16 + (lane & 15);
    const int k0   = kt * 32 + ((lane < 16) ? 0 : 8) + ((j >= 4) ? 16 : 0) + 2 * (j & 3);
    const float a = (k0     < Ksrc) ? src[(size_t)k0 * HDIM + n]       : 0.0f;
    const float b = (k0 + 1 < Ksrc) ? src[(size_t)(k0 + 1) * HDIM + n] : 0.0f;
    dst[id] = pack2h(a, b);
}

// h0 = h_in @ emb_W + emb_b   (N x 16 -> N x 128), trivially memory bound.
__global__ __launch_bounds__(HDIM) void embed_kernel(
        const float* __restrict__ hin, const float* __restrict__ W,
        const float* __restrict__ b, float* __restrict__ hout) {
    const int node = blockIdx.x;
    const int t    = threadIdx.x;
    float acc = b[t];
#pragma unroll
    for (int k = 0; k < 16; ++k)
        acc += hin[node * 16 + k] * W[k * HDIM + t];
    hout[node * HDIM + t] = acc;
}

// ---------------- edge kernel (WMMA-heavy) ----------------
// One block = 16 edges, 128 threads = 4 waves; wave w owns cols [32w, 32w+32).
__global__ __launch_bounds__(128) void edge_kernel(
        const float* __restrict__ equ,  const float* __restrict__ h,
        const float* __restrict__ efea,
        const int* __restrict__ row, const int* __restrict__ col,
        const unsigned* __restrict__ W1sw, const float* __restrict__ b1,
        const unsigned* __restrict__ W2sw, const float* __restrict__ b2,
        const unsigned* __restrict__ cW1sw, const float* __restrict__ cb1,
        const float* __restrict__ cW2,  const float* __restrict__ cb2p,
        float* __restrict__ msg_sum, float* __restrict__ fsum,
        float* __restrict__ cnt) {
    __shared__ unsigned short sA[TILE * KPAD_MSG];  // s_in, f16
    __shared__ unsigned short sH[TILE * HDIM];      // hidden / chidden, f16
    __shared__ float          sMsg[TILE * HDIM];    // message, f32
    __shared__ float          sRij[TILE * 9];
    __shared__ int            sRow[TILE];
    __shared__ float          sRed[TILE * 8];
    __shared__ float          sCm[TILE];

    const int tid  = threadIdx.x;
    const int lane = tid & 31;
    const int wid  = tid >> 5;
    const int e    = tid >> 3;      // local edge 0..15
    const int part = tid & 7;       // 8 threads cooperate per edge
    const int eg   = blockIdx.x * TILE + e;

    const int r = row[eg];
    const int c = col[eg];
    // ---- stage s_in = [scalar(3) | h_row(128) | h_col(128) | ef(8) | pad] ----
#pragma unroll
    for (int i = 0; i < 16; ++i) {
        sA[e * KPAD_MSG + 3 + part * 16 + i]   = f2h(h[r * HDIM + part * 16 + i]);
        sA[e * KPAD_MSG + 131 + part * 16 + i] = f2h(h[c * HDIM + part * 16 + i]);
    }
    if (part == 0) {
        sRow[e] = r;
        float sc0 = 0.f, sc1 = 0.f, sc2 = 0.f;
#pragma unroll
        for (int d = 0; d < 3; ++d) {
            float v0 = equ[r * 9 + d * 3 + 0] - equ[c * 9 + d * 3 + 0];
            float v1 = equ[r * 9 + d * 3 + 1] - equ[c * 9 + d * 3 + 1];
            float v2 = equ[r * 9 + d * 3 + 2] - equ[c * 9 + d * 3 + 2];
            sRij[e * 9 + d * 3 + 0] = v0;
            sRij[e * 9 + d * 3 + 1] = v1;
            sRij[e * 9 + d * 3 + 2] = v2;
            sc0 += v0 * v0; sc1 += v1 * v1; sc2 += v2 * v2;
        }
        sA[e * KPAD_MSG + 0] = f2h(sqrtf(sc0));
        sA[e * KPAD_MSG + 1] = f2h(sqrtf(sc1));
        sA[e * KPAD_MSG + 2] = f2h(sqrtf(sc2));
#pragma unroll
        for (int j = 0; j < 8; ++j)
            sA[e * KPAD_MSG + 259 + j] = f2h(efea[eg * 8 + j]);
        for (int j = 267; j < KPAD_MSG; ++j) sA[e * KPAD_MSG + j] = 0;
    }
    __syncthreads();

    // ---- GEMM1: [16x288] @ [288x128], SiLU -> sH ----
    {
        v8f c0 = {}, c1 = {};
#pragma unroll
        for (int kc = 0; kc < KPAD_MSG / 32; ++kc) {
            v16h a  = loadAfrag(sA, KPAD_MSG, kc, lane);
            v16h bm0 = loadBfrag(W1sw, 8, kc, wid * 2 + 0, lane);
            v16h bm1 = loadBfrag(W1sw, 8, kc, wid * 2 + 1, lane);
            c0 = __builtin_amdgcn_wmma_f32_16x16x32_f16(false, a, false, bm0, (short)0, c0, false, false);
            c1 = __builtin_amdgcn_wmma_f32_16x16x32_f16(false, a, false, bm1, (short)0, c1, false, false);
        }
        storeC_h16(sH, HDIM, wid * 32,      lane, c0, b1, true);
        storeC_h16(sH, HDIM, wid * 32 + 16, lane, c1, b1, true);
    }
    __syncthreads();

    // ---- GEMM2: message = SiLU(hidden @ W2 + b2) -> sMsg (f32) ----
    {
        v8f c0 = {}, c1 = {};
#pragma unroll
        for (int kc = 0; kc < HDIM / 32; ++kc) {
            v16h a  = loadAfrag(sH, HDIM, kc, lane);
            v16h bm0 = loadBfrag(W2sw, 8, kc, wid * 2 + 0, lane);
            v16h bm1 = loadBfrag(W2sw, 8, kc, wid * 2 + 1, lane);
            c0 = __builtin_amdgcn_wmma_f32_16x16x32_f16(false, a, false, bm0, (short)0, c0, false, false);
            c1 = __builtin_amdgcn_wmma_f32_16x16x32_f16(false, a, false, bm1, (short)0, c1, false, false);
        }
        storeC_f32(sMsg, HDIM, wid * 32,      lane, c0, b2, true);
        storeC_f32(sMsg, HDIM, wid * 32 + 16, lane, c1, b2, true);
    }
    __syncthreads();

    // ---- coord GEMM: chidden = SiLU(message @ cW1 + cb1) -> sH ----
    {
        v8f c0 = {}, c1 = {};
#pragma unroll
        for (int kc = 0; kc < HDIM / 32; ++kc) {
            v16h a  = loadAfrag_f32(sMsg, HDIM, kc, lane);
            v16h bm0 = loadBfrag(cW1sw, 8, kc, wid * 2 + 0, lane);
            v16h bm1 = loadBfrag(cW1sw, 8, kc, wid * 2 + 1, lane);
            c0 = __builtin_amdgcn_wmma_f32_16x16x32_f16(false, a, false, bm0, (short)0, c0, false, false);
            c1 = __builtin_amdgcn_wmma_f32_16x16x32_f16(false, a, false, bm1, (short)0, c1, false, false);
        }
        storeC_h16(sH, HDIM, wid * 32,      lane, c0, cb1, true);
        storeC_h16(sH, HDIM, wid * 32 + 16, lane, c1, cb1, true);
    }
    __syncthreads();

    // ---- coord layer 2: coord_msg[e] = chidden[e] . cW2 + cb2 ----
    {
        float p = 0.0f;
#pragma unroll
        for (int i = 0; i < 16; ++i)
            p += h2f(sH[e * HDIM + part * 16 + i]) * cW2[part * 16 + i];
        sRed[e * 8 + part] = p;
    }
    __syncthreads();
    if (tid < TILE) {
        float s = cb2p[0];
#pragma unroll
        for (int j = 0; j < 8; ++j) s += sRed[tid * 8 + j];
        sCm[tid] = s;
    }
    __syncthreads();

    // ---- scatter: sum(message), sum(rij*cm), count ----
#pragma unroll 4
    for (int ee = 0; ee < TILE; ++ee)
        atomicAdd(&msg_sum[(size_t)sRow[ee] * HDIM + tid], sMsg[ee * HDIM + tid]);
    if (tid < TILE) {
        const int rr = sRow[tid];
        const float cm = sCm[tid];
#pragma unroll
        for (int d = 0; d < 9; ++d)
            atomicAdd(&fsum[rr * 9 + d], sRij[tid * 9 + d] * cm);
        atomicAdd(&cnt[rr], 1.0f);
    }
}

// ---------------- neq gate + equ update ----------------
__global__ __launch_bounds__(HDIM) void gate_kernel(
        const float* __restrict__ h,
        const float* __restrict__ W1, const float* __restrict__ b1,
        const float* __restrict__ W2, const float* __restrict__ b2p,
        const float* __restrict__ fsum, const float* __restrict__ cnt,
        float* __restrict__ equ) {
    __shared__ float hL[HDIM];
    __shared__ float red[HDIM];
    __shared__ float gateS;
    const int node = blockIdx.x;
    const int t    = threadIdx.x;
    hL[t] = h[node * HDIM + t];
    __syncthreads();
    float acc = b1[t];
    for (int k = 0; k < HDIM; ++k)
        acc += hL[k] * W1[k * HDIM + t];
    red[t] = silu_f(acc) * W2[t];
    __syncthreads();
    for (int s = HDIM / 2; s > 0; s >>= 1) {
        if (t < s) red[t] += red[t + s];
        __syncthreads();
    }
    if (t == 0) gateS = red[0] + b2p[0];
    __syncthreads();
    if (t < 9) {
        float tf = fsum[node * 9 + t] / fmaxf(cnt[node], 1.0f);
        tf = fminf(fmaxf(tf, -100.0f), 100.0f);
        equ[node * 9 + t] = gateS * equ[node * 9 + t] + tf;
    }
}

// ---------------- node MLP (WMMA) ----------------
// h_next = (SiLU(concat(h, msg_sum) @ nW1 + nb1)) @ nW2 + nb2
__global__ __launch_bounds__(128) void node_kernel(
        const float* __restrict__ h, const float* __restrict__ msum,
        const unsigned* __restrict__ W1sw, const float* __restrict__ b1,
        const unsigned* __restrict__ W2sw, const float* __restrict__ b2,
        float* __restrict__ hout) {
    __shared__ unsigned short sA[TILE * 256];
    __shared__ unsigned short sH[TILE * HDIM];
    const int tid  = threadIdx.x;
    const int lane = tid & 31;
    const int wid  = tid >> 5;
    const int e    = tid >> 3;
    const int part = tid & 7;
    const int node = blockIdx.x * TILE + e;

#pragma unroll
    for (int i = 0; i < 16; ++i) {
        sA[e * 256 + part * 16 + i]       = f2h(h[node * HDIM + part * 16 + i]);
        sA[e * 256 + 128 + part * 16 + i] = f2h(msum[node * HDIM + part * 16 + i]);
    }
    __syncthreads();
    {
        v8f c0 = {}, c1 = {};
#pragma unroll
        for (int kc = 0; kc < 256 / 32; ++kc) {
            v16h a  = loadAfrag(sA, 256, kc, lane);
            v16h bm0 = loadBfrag(W1sw, 8, kc, wid * 2 + 0, lane);
            v16h bm1 = loadBfrag(W1sw, 8, kc, wid * 2 + 1, lane);
            c0 = __builtin_amdgcn_wmma_f32_16x16x32_f16(false, a, false, bm0, (short)0, c0, false, false);
            c1 = __builtin_amdgcn_wmma_f32_16x16x32_f16(false, a, false, bm1, (short)0, c1, false, false);
        }
        storeC_h16(sH, HDIM, wid * 32,      lane, c0, b1, true);
        storeC_h16(sH, HDIM, wid * 32 + 16, lane, c1, b1, true);
    }
    __syncthreads();
    {
        v8f c0 = {}, c1 = {};
#pragma unroll
        for (int kc = 0; kc < HDIM / 32; ++kc) {
            v16h a  = loadAfrag(sH, HDIM, kc, lane);
            v16h bm0 = loadBfrag(W2sw, 8, kc, wid * 2 + 0, lane);
            v16h bm1 = loadBfrag(W2sw, 8, kc, wid * 2 + 1, lane);
            c0 = __builtin_amdgcn_wmma_f32_16x16x32_f16(false, a, false, bm0, (short)0, c0, false, false);
            c1 = __builtin_amdgcn_wmma_f32_16x16x32_f16(false, a, false, bm1, (short)0, c1, false, false);
        }
        const int mbase = (lane < 16) ? 0 : 8;
        const int n0 = wid * 32 + (lane & 15);
#pragma unroll
        for (int j = 0; j < 8; ++j) {
            const int m = mbase + j;
            hout[(size_t)(blockIdx.x * TILE + m) * HDIM + n0]      = c0[j] + b2[n0];
            hout[(size_t)(blockIdx.x * TILE + m) * HDIM + n0 + 16] = c1[j] + b2[n0 + 16];
        }
    }
}

// ---------------- host orchestration ----------------

extern "C" void kernel_launch(void* const* d_in, const int* in_sizes, int n_in,
                              void* d_out, int out_size, void* d_ws, size_t ws_size,
                              hipStream_t stream) {
    const float* equ_in   = (const float*)d_in[0];
    const float* h_in     = (const float*)d_in[1];
    const float* edge_fea = (const float*)d_in[2];
    const float* emb_W    = (const float*)d_in[3];
    const float* emb_b    = (const float*)d_in[4];
    const float* msg_W1   = (const float*)d_in[5];
    const float* msg_b1   = (const float*)d_in[6];
    const float* msg_W2   = (const float*)d_in[7];
    const float* msg_b2   = (const float*)d_in[8];
    const float* coord_W1 = (const float*)d_in[9];
    const float* coord_b1 = (const float*)d_in[10];
    const float* coord_W2 = (const float*)d_in[11];
    const float* coord_b2 = (const float*)d_in[12];
    const float* node_W1  = (const float*)d_in[13];
    const float* node_b1  = (const float*)d_in[14];
    const float* node_W2  = (const float*)d_in[15];
    const float* node_b2  = (const float*)d_in[16];
    const float* neq_W1   = (const float*)d_in[17];
    const float* neq_b1   = (const float*)d_in[18];
    const float* neq_W2   = (const float*)d_in[19];
    const float* neq_b2   = (const float*)d_in[20];
    const int*   eidx     = (const int*)d_in[21];
    const int* row = eidx;
    const int* col = eidx + EE;

    // carve workspace
    char* base = (char*)d_ws;
    size_t cur = 0;
    auto alloc = [&](size_t bytes) -> void* {
        void* p = base + cur;
        cur = (cur + bytes + 255) & ~(size_t)255;
        return p;
    };
    float* equ  = (float*)alloc((size_t)NN * 9 * 4);
    float* h0   = (float*)alloc((size_t)NN * HDIM * 4);
    float* h1   = (float*)alloc((size_t)NN * HDIM * 4);
    float* msum = (float*)alloc((size_t)NN * HDIM * 4);
    float* fsum = (float*)alloc((size_t)NN * 9 * 4);
    float* cnt  = (float*)alloc((size_t)NN * 4);
    unsigned *mW1s[2], *mW2s[2], *cW1s[2], *nW1s[2], *nW2s[2];
    for (int i = 0; i < 2; ++i) {
        mW1s[i] = (unsigned*)alloc((size_t)KPAD_MSG * HDIM * 2);
        mW2s[i] = (unsigned*)alloc((size_t)HDIM * HDIM * 2);
        cW1s[i] = (unsigned*)alloc((size_t)HDIM * HDIM * 2);
        nW1s[i] = (unsigned*)alloc((size_t)256 * HDIM * 2);
        nW2s[i] = (unsigned*)alloc((size_t)HDIM * HDIM * 2);
    }

    // prep: copy equ, embed h, swizzle weights to f16 B-fragment layout
    copy_f32<<<(NN * 9 + 255) / 256, 256, 0, stream>>>(equ_in, equ, NN * 9);
    embed_kernel<<<NN, HDIM, 0, stream>>>(h_in, emb_W, emb_b, h0);
    for (int i = 0; i < 2; ++i) {
        swizzle_w_kernel<<<(KPAD_MSG * 64 + 255) / 256, 256, 0, stream>>>(
            msg_W1 + (size_t)i * 267 * HDIM, 267, KPAD_MSG, mW1s[i]);
        swizzle_w_kernel<<<(HDIM * 64 + 255) / 256, 256, 0, stream>>>(
            msg_W2 + (size_t)i * HDIM * HDIM, HDIM, HDIM, mW2s[i]);
        swizzle_w_kernel<<<(HDIM * 64 + 255) / 256, 256, 0, stream>>>(
            coord_W1 + (size_t)i * HDIM * HDIM, HDIM, HDIM, cW1s[i]);
        swizzle_w_kernel<<<(256 * 64 + 255) / 256, 256, 0, stream>>>(
            node_W1 + (size_t)i * 256 * HDIM, 256, 256, nW1s[i]);
        swizzle_w_kernel<<<(HDIM * 64 + 255) / 256, 256, 0, stream>>>(
            node_W2 + (size_t)i * HDIM * HDIM, HDIM, HDIM, nW2s[i]);
    }

    float* hcur = h0;
    float* hnext = h1;
    for (int i = 0; i < 2; ++i) {
        fill_f32<<<(NN * HDIM + 255) / 256, 256, 0, stream>>>(msum, NN * HDIM);
        fill_f32<<<(NN * 9 + 255) / 256, 256, 0, stream>>>(fsum, NN * 9);
        fill_f32<<<(NN + 255) / 256, 256, 0, stream>>>(cnt, NN);

        edge_kernel<<<EE / TILE, 128, 0, stream>>>(
            equ, hcur, edge_fea, row, col,
            mW1s[i], msg_b1 + i * HDIM,
            mW2s[i], msg_b2 + i * HDIM,
            cW1s[i], coord_b1 + i * HDIM,
            coord_W2 + i * HDIM, coord_b2 + i,
            msum, fsum, cnt);

        gate_kernel<<<NN, HDIM, 0, stream>>>(
            hcur, neq_W1 + (size_t)i * HDIM * HDIM, neq_b1 + i * HDIM,
            neq_W2 + i * HDIM, neq_b2 + i, fsum, cnt, equ);

        node_kernel<<<NN / TILE, 128, 0, stream>>>(
            hcur, msum, nW1s[i], node_b1 + i * HDIM,
            nW2s[i], node_b2 + i * HDIM, hnext);

        float* t = hcur; hcur = hnext; hnext = t;
    }

    float* out = (float*)d_out;
    copy_f32<<<(NN * 9 + 255) / 256, 256, 0, stream>>>(equ, out, NN * 9);
    copy_f32<<<(NN * HDIM + 255) / 256, 256, 0, stream>>>(hcur, out + NN * 9, NN * HDIM);
}
